// gconv_lstm_13975823581349
// MI455X (gfx1250) — compile-verified, matrix operand order checked by reading, not run on hardware
//
#include <hip/hip_runtime.h>
#include <stdint.h>

// Problem constants from the reference (fixed shapes)
#define MDIM 8192   // B*T = 16*512
#define NDIM 2048   // output nodes (j)
#define KDIM 2048   // reduction nodes (k)
#define UDIM 64     // units

typedef _Float16 v8h  __attribute__((ext_vector_type(8)));
typedef _Float16 v16h __attribute__((ext_vector_type(16)));
typedef float    v8f  __attribute__((ext_vector_type(8)));

// ---------------------------------------------------------------------------
// Workspace layout (d_ws):
//   [0      , 8192 )  : w_bar  float[2048]
//   [8192   , 8196 )  : bmean  float
//   [16384  , +8MB )  : Bm     _Float16[2048*2048]  (row j, contiguous k)
// ---------------------------------------------------------------------------

// Kernel 1: column means of W and mean of b
__global__ __launch_bounds__(256) void prep_wbar_kernel(
    const float* __restrict__ W, const float* __restrict__ bias,
    float* __restrict__ w_bar, float* __restrict__ bmean) {
  const int k = blockIdx.x * 256 + threadIdx.x;   // grid sized to exactly KDIM
  float s = 0.0f;
#pragma unroll 8
  for (int i = 0; i < UDIM; ++i) s += W[i * KDIM + k];
  w_bar[k] = s * (1.0f / UDIM);
  if (blockIdx.x == 0 && threadIdx.x == 0) {
    float sb = 0.0f;
    for (int i = 0; i < UDIM; ++i) sb += bias[i];
    *bmean = sb * (1.0f / UDIM);
  }
}

// Kernel 2: Bm[j,k] = f16(Adj[j,k] * w_bar[k]); 8 elements per thread
__global__ __launch_bounds__(256) void prep_bmat_kernel(
    const float* __restrict__ Adj, const float* __restrict__ w_bar,
    _Float16* __restrict__ Bm) {
  const long e = ((long)blockIdx.x * 256 + (long)threadIdx.x) * 8;
  const int k = (int)(e & (long)(KDIM - 1));
  const float4 a0 = *(const float4*)(Adj + e);
  const float4 a1 = *(const float4*)(Adj + e + 4);
  const float4 w0 = *(const float4*)(w_bar + k);
  const float4 w1 = *(const float4*)(w_bar + k + 4);
  v8h h;
  h[0] = (_Float16)(a0.x * w0.x);
  h[1] = (_Float16)(a0.y * w0.y);
  h[2] = (_Float16)(a0.z * w0.z);
  h[3] = (_Float16)(a0.w * w0.w);
  h[4] = (_Float16)(a1.x * w1.x);
  h[5] = (_Float16)(a1.y * w1.y);
  h[6] = (_Float16)(a1.z * w1.z);
  h[7] = (_Float16)(a1.w * w1.w);
  *(v8h*)(Bm + e) = h;
}

// ---------------------------------------------------------------------------
// Kernel 3: GEMM out = X * Bm^T + bmean via v_wmma_f32_16x16x32_f16
// Block: 256 threads (8 waves). Tile: 128(M) x 128(N), K-step 32.
// Wave grid 4(M) x 2(N); each wave: 2x4 WMMA tiles of 16x16.
// A (f32 x) staged through VGPRs with f32->f16 cvt; B (already f16) staged
// with GLOBAL_LOAD_ASYNC_TO_LDS_B128 on the async path (ASYNCcnt).
// ---------------------------------------------------------------------------

#define KT   32   // K per LDS stage (one WMMA depth)
#define LDST 40   // LDS row stride in halves (32 + 8 pad; 80B keeps 16B align)

// Load one 16x32 f16 fragment from LDS per ISA layout:
//   lane<16: row = ll, K chunks {0..7, 16..23}; lane>=16: row = ll, {8..15, 24..31}
__device__ __forceinline__ v16h ld_frag(const _Float16* p) {
  const v8h lo = *(const v8h*)(p);        // K = base .. base+7
  const v8h hi = *(const v8h*)(p + 16);   // K = base+16 .. base+23
  return __builtin_shufflevector(lo, hi, 0, 1, 2, 3, 4, 5, 6, 7,
                                 8, 9, 10, 11, 12, 13, 14, 15);
}

// Convert 16 f32 (4x float4) to f16 and store 32B into LDS
__device__ __forceinline__ void stage_storeA(_Float16* da, const float4* fa) {
  v8h h0, h1;
  h0[0] = (_Float16)fa[0].x; h0[1] = (_Float16)fa[0].y;
  h0[2] = (_Float16)fa[0].z; h0[3] = (_Float16)fa[0].w;
  h0[4] = (_Float16)fa[1].x; h0[5] = (_Float16)fa[1].y;
  h0[6] = (_Float16)fa[1].z; h0[7] = (_Float16)fa[1].w;
  h1[0] = (_Float16)fa[2].x; h1[1] = (_Float16)fa[2].y;
  h1[2] = (_Float16)fa[2].z; h1[3] = (_Float16)fa[2].w;
  h1[4] = (_Float16)fa[3].x; h1[5] = (_Float16)fa[3].y;
  h1[6] = (_Float16)fa[3].z; h1[7] = (_Float16)fa[3].w;
  *(v8h*)(da)     = h0;
  *(v8h*)(da + 8) = h1;
}

// Async copy of one 32B B-row segment: global (gaddr) -> LDS (lds byte offset).
// INST_OFFSET is added to BOTH global and LDS addresses, so offset:16 covers
// the second 16B chunk on both sides. Tracked by ASYNCcnt.
__device__ __forceinline__ void async_copy_B32B(uint32_t lds_off, const char* gaddr) {
  asm volatile("global_load_async_to_lds_b128 %0, %1, off"
               :: "v"(lds_off), "v"(gaddr) : "memory");
  asm volatile("global_load_async_to_lds_b128 %0, %1, off offset:16"
               :: "v"(lds_off), "v"(gaddr) : "memory");
}

__device__ __forceinline__ void wait_async0() {
  asm volatile("s_wait_asynccnt 0x0" ::: "memory");
}

__global__ __launch_bounds__(256) void gemm_wmma_kernel(
    const float* __restrict__ X, const _Float16* __restrict__ Bm,
    const float* __restrict__ bmeanp, float* __restrict__ Out) {
  __shared__ alignas(16) _Float16 As[2][128 * LDST];
  __shared__ alignas(16) _Float16 Bs[2][128 * LDST];

  const int tid  = threadIdx.x;
  const int lane = tid & 31;      // wave32
  const int wave = tid >> 5;      // 0..7
  const int wm   = wave & 3;      // wave M index (0..3) -> 32 rows each
  const int wn   = wave >> 2;     // wave N index (0..1) -> 64 cols each
  const int ll   = lane & 15;
  const int lh   = lane >> 4;

  const int m0 = blockIdx.y * 128;
  const int j0 = blockIdx.x * 128;

  // global->LDS staging map: 2 threads per tile row, 16 elements each
  const int grow = tid >> 1;
  const int gcol = (tid & 1) * 16;

  const float* pA = X + (long)(m0 + grow) * KDIM + gcol;
  const char*  gB = (const char*)(Bm + (long)(j0 + grow) * KDIM + gcol);

  // Wave-relative LDS byte offsets for the async B destination (flat-aperture
  // rule: low 32 bits of a generic LDS pointer are the LDS byte address).
  const uint32_t ldsB0 = (uint32_t)(uintptr_t)&Bs[0][grow * LDST + gcol];
  const uint32_t ldsB1 = (uint32_t)(uintptr_t)&Bs[1][grow * LDST + gcol];

  float4 fa[4];
  v8f    acc[2][4] = {};

  // Prologue: async B tile 0 -> Bs[0]; A tile 0 via regs -> As[0]
  async_copy_B32B(ldsB0, gB);
#pragma unroll
  for (int q = 0; q < 4; ++q) fa[q] = *(const float4*)(pA + q * 4);
  stage_storeA(&As[0][grow * LDST + gcol], fa);

  const int nkt = KDIM / KT;   // 64
  for (int kt = 0; kt < nkt; ++kt) {
    // Own-wave async loads (into buf kt&1, issued last iteration) must land
    // before we signal the barrier; the barrier then publishes across waves.
    wait_async0();
    __syncthreads();
    const int  buf  = kt & 1;
    const bool more = (kt + 1) < nkt;

    // Kick next K-tile immediately after the barrier: async B straight to
    // LDS, A into VGPRs (both overlap the WMMA block below).
    if (more) {
      pA += KT;
      gB += KT * (int)sizeof(_Float16);
      async_copy_B32B(buf ? ldsB0 : ldsB1, gB);
#pragma unroll
      for (int q = 0; q < 4; ++q) fa[q] = *(const float4*)(pA + q * 4);
    }

    v16h a[2], b[4];
#pragma unroll
    for (int i = 0; i < 2; ++i)
      a[i] = ld_frag(&As[buf][(wm * 32 + i * 16 + ll) * LDST + lh * 8]);
#pragma unroll
    for (int n = 0; n < 4; ++n)
      b[n] = ld_frag(&Bs[buf][(wn * 64 + n * 16 + ll) * LDST + lh * 8]);

#pragma unroll
    for (int i = 0; i < 2; ++i)
#pragma unroll
      for (int n = 0; n < 4; ++n)
        acc[i][n] = __builtin_amdgcn_wmma_f32_16x16x32_f16(
            false, a[i], false, b[n], (short)0, acc[i][n], false, false);

    // Convert + stage A's next tile into the other buffer (safe: nobody reads
    // it until the barrier at the top of the next iteration).
    if (more)
      stage_storeA(&As[buf ^ 1][grow * LDST + gcol], fa);
  }

  // Epilogue: C/D layout — VGPR v: lanes 0-15 -> M=v, lanes 16-31 -> M=v+8
  const float bmean = *bmeanp;
#pragma unroll
  for (int i = 0; i < 2; ++i) {
    const int mrow = m0 + wm * 32 + i * 16 + lh * 8;
#pragma unroll
    for (int n = 0; n < 4; ++n) {
      const int col = j0 + wn * 64 + n * 16 + ll;
      float* po = Out + (long)mrow * NDIM + col;
#pragma unroll
      for (int v = 0; v < 8; ++v)
        po[(long)v * NDIM] = acc[i][n][v] + bmean;
    }
  }
}

// ---------------------------------------------------------------------------
extern "C" void kernel_launch(void* const* d_in, const int* in_sizes, int n_in,
                              void* d_out, int out_size, void* d_ws, size_t ws_size,
                              hipStream_t stream) {
  const float* x   = (const float*)d_in[0];   // [16,512,2048] f32
  const float* Adj = (const float*)d_in[1];   // [2048,2048]   f32
  const float* W   = (const float*)d_in[2];   // [64,2048]     f32
  const float* b   = (const float*)d_in[3];   // [64]          f32
  float*       out = (float*)d_out;           // [16,512,2048] f32

  float*    w_bar = (float*)d_ws;
  float*    bmean = (float*)((char*)d_ws + 8192);
  _Float16* Bm    = (_Float16*)((char*)d_ws + 16384);  // 8 MB

  // 1) column mean of W (-> w_bar) and mean of b (-> bmean)
  prep_wbar_kernel<<<KDIM / 256, 256, 0, stream>>>(W, b, w_bar, bmean);

  // 2) effective B matrix in f16: Bm[j,k] = Adj[j,k] * w_bar[k]
  prep_bmat_kernel<<<(NDIM * KDIM / 8) / 256, 256, 0, stream>>>(Adj, w_bar, Bm);

  // 3) 8192x2048x2048 GEMM on the WMMA pipes, +bmean fused into the store
  dim3 grid(NDIM / 128, MDIM / 128);   // (16, 64)
  gemm_wmma_kernel<<<grid, 256, 0, stream>>>(x, Bm, bmean, out);
}
